// NGPNerf_16277926052174
// MI455X (gfx1250) — matrix operand order repaired and checked by reading, not compiled, not run on hardware
//
#include <hip/hip_runtime.h>
#include <math.h>

typedef __attribute__((ext_vector_type(16))) _Float16 v16h;
typedef __attribute__((ext_vector_type(8)))  _Float16 v8h;
typedef __attribute__((ext_vector_type(8)))  float    v8f;

#define NS   262144
#define LV   16
#define TBL  524288u      // 2^19 hash entries per level (power of two -> mask)
#define WAVES 8
#define TILES 4
#define BLOCK (WAVES * 32)
#define SPB   (WAVES * 16 * TILES)   // samples per block = 512
#define NTILE 20                     // WMMA B tiles resident in LDS

struct Scales { float s[LV]; };

// Intra-wave ordering fence: H/C tiles are wave-private; wave32 lockstep +
// in-order LDS make cross-lane store->load safe once the compiler keeps order.
#define WAVE_FENCE() __builtin_amdgcn_wave_barrier()

__device__ __forceinline__ v8f wmma_f16(v16h a, v16h b, v8f c) {
    return __builtin_amdgcn_wmma_f32_16x16x32_f16(false, a, false, b, (short)0, c, false, false);
}

// Pre-swizzled B fragment: 32 bytes contiguous per lane -> 2x ds_load_b128.
__device__ __forceinline__ v16h load_bw(const _Float16* WB, int tile, int lane) {
    return *(const v16h*)(WB + ((tile * 32 + lane) << 4));
}

// A fragment from a contiguous activation row (logical feature order):
// positions 0..7 = row[k0+8g .. +7], positions 8..15 = row[k0+8g+16 .. +23].
__device__ __forceinline__ v16h load_a_frag(const _Float16* row, int k0, int g) {
    const v8h lo = *(const v8h*)(row + k0 + 8 * g);
    const v8h hi = *(const v8h*)(row + k0 + 8 * g + 16);
    return __builtin_shufflevector(lo, hi, 0, 1, 2, 3, 4, 5, 6, 7,
                                           8, 9, 10, 11, 12, 13, 14, 15);
}

__global__ __launch_bounds__(BLOCK) void ngp_fused(
    const float* __restrict__ pos, const float* __restrict__ dir,
    const float* __restrict__ table,
    const float* __restrict__ w1, const float* __restrict__ b1,
    const float* __restrict__ w2, const float* __restrict__ b2,
    const float* __restrict__ w3, const float* __restrict__ b3,
    const float* __restrict__ w4, const float* __restrict__ b4,
    const float* __restrict__ w5, const float* __restrict__ b5,
    float* __restrict__ out, Scales sc)
{
    // B tiles pre-swizzled into per-lane fragment order:
    //  t0..3 : W1 (k0=0,   n0=16t)     t4..5 : W2 (k0=32c, n0=0)
    //  t6..9 : W3 (k0=0,   n0=16(t-6)) t10..17: W4 (to=(t-10)/2, k0=32((t-10)&1))
    //  t18..19: W5 (k0=32(t-18), n0=0, cols 3..15 zero)
    __shared__ __align__(32) _Float16 WB[NTILE * 32 * 16];
    __shared__ float B1[64], B2[16], B3[64], B4[64], B5[4];
    __shared__ __align__(16) _Float16 H[WAVES][16][72];   // per-wave activations (144B rows)
    __shared__ __align__(16) _Float16 C[WAVES][16][40];   // per-wave concat tile (80B rows)

    const int tid = threadIdx.x;
    for (int i = tid; i < NTILE * 32 * 16; i += BLOCK) {
        const int t  = i >> 9;          // tile
        const int l2 = (i >> 4) & 31;   // lane slot
        const int e  = i & 15;          // element within fragment
        const int gg = l2 >> 4, nn = l2 & 15;
        const int kk = 8 * gg + (e & 7) + ((e & 8) << 1);
        float v;
        if (t < 4) {                         // W1: 32x64
            const int k = kk, n = 16 * t + nn;
            v = w1[k * 64 + n];
        } else if (t < 6) {                  // W2: 64x16
            const int k = 32 * (t - 4) + kk;
            v = w2[k * 16 + nn];
        } else if (t < 10) {                 // W3: 32x64
            const int k = kk, n = 16 * (t - 6) + nn;
            v = w3[k * 64 + n];
        } else if (t < 18) {                 // W4: 64x64
            const int to = (t - 10) >> 1, ch = (t - 10) & 1;
            const int k = 32 * ch + kk, n = 16 * to + nn;
            v = w4[k * 64 + n];
        } else {                             // W5: 64x3 zero-padded to 64x16
            const int k = 32 * (t - 18) + kk;
            v = (nn < 3) ? w5[k * 3 + nn] : 0.0f;
        }
        WB[i] = (_Float16)v;
    }
    if (tid < 64) { B1[tid] = b1[tid]; B3[tid] = b3[tid]; B4[tid] = b4[tid]; }
    if (tid < 16) B2[tid] = b2[tid];
    if (tid < 4)  B5[tid] = (tid < 3) ? b5[tid] : 0.0f;
    __syncthreads();   // only cross-wave barrier: weights/biases now resident

    const int wave = tid >> 5;
    const int lane = tid & 31;
    const int m    = lane & 15;      // sample-in-tile / N column
    const int g    = lane >> 4;      // lane group
    const float2* __restrict__ tbl = (const float2*)table;

    for (int t = 0; t < TILES; ++t) {
        const int base   = blockIdx.x * SPB + wave * (16 * TILES) + t * 16;
        const int sample = base + m;

        // ---------- hash encoding: lane computes 8 levels of its sample ----------
        // level assignment matches WMMA A physical-K layout:
        // g=0 -> levels {0..3, 8..11}, g=1 -> levels {4..7, 12..15}
        const float px = (pos[sample * 3 + 0] + 1.0f) * 0.5f;
        const float py = (pos[sample * 3 + 1] + 1.0f) * 0.5f;
        const float pz = (pos[sample * 3 + 2] + 1.0f) * 0.5f;
        v16h afeat;
#pragma unroll
        for (int j = 0; j < 8; ++j) {
            const int lvl = ((j < 4) ? j : j + 4) + 4 * g;
            const float s = sc.s[lvl];
            const float sx = px * s, sy = py * s, sz = pz * s;
            const float flx = floorf(sx), fly = floorf(sy), flz = floorf(sz);
            const float frx = sx - flx, fry = sy - fly, frz = sz - flz;
            const unsigned ix = (unsigned)flx, iy = (unsigned)fly, iz = (unsigned)flz;
            // precompute per-axis hash terms; corner loop is XOR-only
            const unsigned hx0 = ix,                    hx1 = ix + 1u;
            const unsigned hy0 = iy * 2654435761u,      hy1 = hy0 + 2654435761u;
            const unsigned hz0 = iz * 805459861u,       hz1 = hz0 + 805459861u;
            const float wx0 = 1.0f - frx, wy0 = 1.0f - fry, wz0 = 1.0f - frz;
            float ax = 0.0f, ay = 0.0f;
#pragma unroll
            for (int c8 = 0; c8 < 8; ++c8) {
                const unsigned h = ((c8 & 4) ? hx1 : hx0) ^
                                   ((c8 & 2) ? hy1 : hy0) ^
                                   ((c8 & 1) ? hz1 : hz0);
                const unsigned idx = (h & (TBL - 1u)) + (unsigned)lvl * TBL;
                const float2 f = tbl[idx];
                const float w = ((c8 & 4) ? frx : wx0) *
                                ((c8 & 2) ? fry : wy0) *
                                ((c8 & 1) ? frz : wz0);
                ax = fmaf(w, f.x, ax);
                ay = fmaf(w, f.y, ay);
            }
            afeat[2 * j + 0] = (_Float16)ax;
            afeat[2 * j + 1] = (_Float16)ay;
        }

        // ---------- layer 1: (16x32) @ (32x64), ReLU ----------
        v8f h1[4];
#pragma unroll
        for (int to = 0; to < 4; ++to) {
            v8f acc = {};
            h1[to] = wmma_f16(afeat, load_bw(WB, to, lane), acc);
        }
        WAVE_FENCE();   // previous tile's layer-5 H reads ordered before overwrite
#pragma unroll
        for (int to = 0; to < 4; ++to)
#pragma unroll
            for (int j = 0; j < 8; ++j) {
                const int row = j + 8 * g, col = to * 16 + m;
                const float v = h1[to][j] + B1[col];
                H[wave][row][col] = (_Float16)fmaxf(v, 0.0f);
            }
        WAVE_FENCE();

        // ---------- layer 2: (16x64) @ (64x16); density = exp(col0) ----------
        v8f h2 = {};
        h2 = wmma_f16(load_a_frag(&H[wave][m][0], 0,  g), load_bw(WB, 4, lane), h2);
        h2 = wmma_f16(load_a_frag(&H[wave][m][0], 32, g), load_bw(WB, 5, lane), h2);
#pragma unroll
        for (int j = 0; j < 8; ++j) {
            const int row = j + 8 * g;
            const float v = h2[j] + B2[m];
            C[wave][row][m] = (_Float16)v;          // raw h (no ReLU) feeds concat
            if (m == 0) out[(size_t)(base + row) * 4 + 0] = expf(v);
        }

        // ---------- SH encode (lanes 0..15, one sample each) ----------
        if (g == 0) {
            const float dx = dir[sample * 3 + 0];
            const float dy = dir[sample * 3 + 1];
            const float dz = dir[sample * 3 + 2];
            const float xx = dx * dx, yy = dy * dy, zz = dz * dz;
            const float xy = dx * dy, yz = dy * dz, xz = dx * dz;
            v8h s0, s1;
            s0[0] = (_Float16)(0.28209479177387814f);
            s0[1] = (_Float16)(-0.48860251190291987f * dy);
            s0[2] = (_Float16)(0.48860251190291987f * dz);
            s0[3] = (_Float16)(-0.48860251190291987f * dx);
            s0[4] = (_Float16)(1.0925484305920792f * xy);
            s0[5] = (_Float16)(-1.0925484305920792f * yz);
            s0[6] = (_Float16)(0.94617469575755997f * zz - 0.31539156525252005f);
            s0[7] = (_Float16)(-1.0925484305920792f * xz);
            s1[0] = (_Float16)(0.54627421529603959f * (xx - yy));
            s1[1] = (_Float16)(0.59004358992664352f * dy * (-3.0f * xx + yy));
            s1[2] = (_Float16)(2.8906114426405538f * xy * dz);
            s1[3] = (_Float16)(0.45704579946446572f * dy * (1.0f - 5.0f * zz));
            s1[4] = (_Float16)(0.3731763325901154f * dz * (5.0f * zz - 3.0f));
            s1[5] = (_Float16)(0.45704579946446572f * dx * (1.0f - 5.0f * zz));
            s1[6] = (_Float16)(1.445305721320277f * dz * (xx - yy));
            s1[7] = (_Float16)(0.59004358992664352f * dx * (xx - 3.0f * yy));
            *(v8h*)&C[wave][m][16] = s0;
            *(v8h*)&C[wave][m][24] = s1;
        }
        WAVE_FENCE();

        // ---------- layer 3: (16x32) @ (32x64), ReLU ----------
        {
            const v16h a3 = load_a_frag(&C[wave][m][0], 0, g);
            v8f h3[4];
#pragma unroll
            for (int to = 0; to < 4; ++to) {
                v8f acc = {};
                h3[to] = wmma_f16(a3, load_bw(WB, 6 + to, lane), acc);
            }
            WAVE_FENCE();   // layer-2 A reads of H ordered before overwrite
#pragma unroll
            for (int to = 0; to < 4; ++to)
#pragma unroll
                for (int j = 0; j < 8; ++j) {
                    const int row = j + 8 * g, col = to * 16 + m;
                    const float v = h3[to][j] + B3[col];
                    H[wave][row][col] = (_Float16)fmaxf(v, 0.0f);
                }
        }
        WAVE_FENCE();

        // ---------- layer 4: (16x64) @ (64x64), ReLU ----------
        {
            const v16h a0 = load_a_frag(&H[wave][m][0], 0,  g);
            const v16h a1 = load_a_frag(&H[wave][m][0], 32, g);
            v8f h4[4];
#pragma unroll
            for (int to = 0; to < 4; ++to) {
                v8f acc = {};
                acc = wmma_f16(a0, load_bw(WB, 10 + 2 * to,     lane), acc);
                acc = wmma_f16(a1, load_bw(WB, 10 + 2 * to + 1, lane), acc);
                h4[to] = acc;
            }
            WAVE_FENCE();   // all A reads of H ordered before overwrite
#pragma unroll
            for (int to = 0; to < 4; ++to)
#pragma unroll
                for (int j = 0; j < 8; ++j) {
                    const int row = j + 8 * g, col = to * 16 + m;
                    const float v = h4[to][j] + B4[col];
                    H[wave][row][col] = (_Float16)fmaxf(v, 0.0f);
                }
        }
        WAVE_FENCE();

        // ---------- layer 5: (16x64) @ (64x16 padded), sigmoid, write rgb ----------
        {
            v8f acc = {};
            acc = wmma_f16(load_a_frag(&H[wave][m][0], 0,  g), load_bw(WB, 18, lane), acc);
            acc = wmma_f16(load_a_frag(&H[wave][m][0], 32, g), load_bw(WB, 19, lane), acc);
            if (m < 3) {
#pragma unroll
                for (int j = 0; j < 8; ++j) {
                    const int row = j + 8 * g;
                    const float v = acc[j] + B5[m];
                    out[(size_t)(base + row) * 4 + 1 + m] = 1.0f / (1.0f + expf(-v));
                }
            }
        }
        WAVE_FENCE();
    }
}

extern "C" void kernel_launch(void* const* d_in, const int* in_sizes, int n_in,
                              void* d_out, int out_size, void* d_ws, size_t ws_size,
                              hipStream_t stream) {
    (void)in_sizes; (void)n_in; (void)out_size; (void)d_ws; (void)ws_size;
    const float* pos  = (const float*)d_in[0];
    const float* dirv = (const float*)d_in[1];
    const float* tab  = (const float*)d_in[2];
    const float* w1 = (const float*)d_in[3];  const float* b1 = (const float*)d_in[4];
    const float* w2 = (const float*)d_in[5];  const float* b2 = (const float*)d_in[6];
    const float* w3 = (const float*)d_in[7];  const float* b3 = (const float*)d_in[8];
    const float* w4 = (const float*)d_in[9];  const float* b4 = (const float*)d_in[10];
    const float* w5 = (const float*)d_in[11]; const float* b5 = (const float*)d_in[12];
    float* out = (float*)d_out;

    Scales sc;
    const double growth = exp((log(2048.0) - log(16.0)) / (LV - 1));
    for (int l = 0; l < LV; ++l) sc.s[l] = (float)floor(16.0 * pow(growth, (double)l));

    const int grid = NS / SPB;   // 512 blocks of 256 threads
    ngp_fused<<<grid, BLOCK, 0, stream>>>(pos, dirv, tab,
                                          w1, b1, w2, b2, w3, b3, w4, b4, w5, b5,
                                          out, sc);
}
